// InstantNGP2D_11201274708327
// MI455X (gfx1250) — matrix operand order repaired and checked by reading, not compiled, not run on hardware
//
#include <hip/hip_runtime.h>
#include <cmath>

typedef __attribute__((ext_vector_type(16))) _Float16 v16h;
typedef __attribute__((ext_vector_type(8)))  _Float16 v8h;
typedef __attribute__((ext_vector_type(8)))  float    v8f;

#define NLEV    16
#define TBL     (1u << 19)
#define TMASK   (TBL - 1u)
#define PRIME_Y 2654435761u

struct LP {
    float    scale[NLEV];
    unsigned res[NLEV];
    unsigned dense_mask;
};

// ---------------------------------------------------------------------------
// Weight prep: f32 [K][N] -> f16 [N][K] (transposed, w2 padded to 16 cols)
// ws layout (halfs): w0T[64][32] @0, w1T[64][64] @2048, w2T[16][64] @6144
// ---------------------------------------------------------------------------
__global__ void prep_weights(const float* __restrict__ w0,
                             const float* __restrict__ w1,
                             const float* __restrict__ w2,
                             _Float16* __restrict__ ws) {
    int i = blockIdx.x * blockDim.x + threadIdx.x;
    if (i < 2048) {                       // w0T: n in [0,64), k in [0,32)
        int n = i >> 5, k = i & 31;
        ws[i] = (_Float16)w0[k * 64 + n];
    } else if (i < 2048 + 4096) {         // w1T: n in [0,64), k in [0,64)
        int j = i - 2048;
        int n = j >> 6, k = j & 63;
        ws[i] = (_Float16)w1[k * 64 + n];
    } else if (i < 2048 + 4096 + 1024) {  // w2T: n in [0,16) (pad), k in [0,64)
        int j = i - 6144;
        int n = j >> 6, k = j & 63;
        ws[i] = (_Float16)(n < 3 ? w2[k * 3 + n] : 0.0f);
    }
}

// ---------------------------------------------------------------------------
// Fragment loaders (wave32, v_wmma_f32_16x16x32_f16 layouts)
// ---------------------------------------------------------------------------
// A: 16x32 f16, rows M striped in-lane. Lane half (lane>=16) -> K base +8.
// elems 0..7  = K kb..kb+7 ; elems 8..15 = K kb+16..kb+23  (kb = (lane>>4)*8)
__device__ __forceinline__ v16h load_frag_a(const _Float16* base, int ldK, int lane) {
    int m  = lane & 15;
    int kb = (lane >> 4) << 3;
    v8h lo = *(const v8h*)(base + m * ldK + kb);
    v8h hi = *(const v8h*)(base + m * ldK + kb + 16);
    v16h a;
#pragma unroll
    for (int i = 0; i < 8; ++i) { a[i] = lo[i]; a[i + 8] = hi[i]; }
    return a;
}

// B: 32x16 f16, column-major per lane: lane holds column N = lane&15,
// elems 0..7 = K kb..kb+7 ; elems 8..15 = K kb+16..kb+23 (kb = (lane>>4)*8).  base points at wT[N][K] chunk start.
__device__ __forceinline__ v16h load_frag_b(const _Float16* base, int ldK, int lane) {
    int n  = lane & 15;
    int kb = (lane >> 4) << 3;
    v8h lo = *(const v8h*)(base + n * ldK + kb);
    v8h hi = *(const v8h*)(base + n * ldK + kb + 16);
    v16h b;
#pragma unroll
    for (int i = 0; i < 8; ++i) { b[i] = lo[i]; b[i + 8] = hi[i]; }
    return b;
}

// ---------------------------------------------------------------------------
// Fused hash-encode + MLP.  128 threads = 4 waves; each wave owns 32 points.
// ---------------------------------------------------------------------------
__launch_bounds__(128)
__global__ void ngp_fused(const float* __restrict__ uv,
                          const float* __restrict__ table,
                          const _Float16* __restrict__ wsT,
                          float* __restrict__ out,
                          LP lp, int Npts) {
    __shared__ _Float16 featL[4][32 * 32];  // 8 KB  (per-wave 32 pts x 32 feats)
    __shared__ _Float16 h0L[4][32 * 64];    // 16 KB
    __shared__ _Float16 h1L[4][32 * 64];    // 16 KB

    const int lane   = threadIdx.x & 31;
    const int wid    = threadIdx.x >> 5;
    const int ptBase = (blockIdx.x * 4 + wid) * 32;
    const int pt     = ptBase + lane;
    const int ptc    = pt < Npts ? pt : Npts - 1;

    // ---------------- hash-grid encoding (1 point per lane) ----------------
    const float u = uv[2 * ptc + 0];
    const float v = uv[2 * ptc + 1];
    _Float16* feat = &featL[wid][0];

#pragma unroll
    for (int l = 0; l < NLEV; ++l) {
        const float    s     = lp.scale[l];
        const unsigned res   = lp.res[l];
        const bool     dense = (lp.dense_mask >> l) & 1u;

        float px = u * s + 0.5f, py = v * s + 0.5f;
        float fpx = floorf(px), fpy = floorf(py);
        float fx = px - fpx,    fy = py - fpy;
        unsigned x0 = (unsigned)fpx, y0 = (unsigned)fpy;
        unsigned x1 = x0 + 1u,       y1 = y0 + 1u;

        unsigned i00, i10, i01, i11;
        if (dense) {
            unsigned r0 = y0 * res, r1 = y1 * res;
            i00 = (x0 + r0) & TMASK; i10 = (x1 + r0) & TMASK;
            i01 = (x0 + r1) & TMASK; i11 = (x1 + r1) & TMASK;
        } else {
            unsigned h0 = y0 * PRIME_Y, h1 = y1 * PRIME_Y;
            i00 = (x0 ^ h0) & TMASK; i10 = (x1 ^ h0) & TMASK;
            i01 = (x0 ^ h1) & TMASK; i11 = (x1 ^ h1) & TMASK;
        }

        const float2* tl = (const float2*)table + (size_t)l * TBL;
        float2 c00 = tl[i00], c10 = tl[i10], c01 = tl[i01], c11 = tl[i11];

        float w00 = (1.f - fx) * (1.f - fy), w10 = fx * (1.f - fy);
        float w01 = (1.f - fx) * fy,         w11 = fx * fy;
        float f0 = w00 * c00.x + w10 * c10.x + w01 * c01.x + w11 * c11.x;
        float f1 = w00 * c00.y + w10 * c10.y + w01 * c01.y + w11 * c11.y;

        feat[lane * 32 + 2 * l + 0] = (_Float16)f0;
        feat[lane * 32 + 2 * l + 1] = (_Float16)f1;
    }
    asm volatile("s_wait_dscnt 0" ::: "memory");  // cross-lane feat visible (same wave, in-order)

    // ------------------------------- MLP -----------------------------------
    const _Float16* w0T = wsT;                  // [64][32]
    const _Float16* w1T = wsT + 64 * 32;        // [64][64]
    const _Float16* w2T = wsT + 64 * 32 + 64 * 64;  // [16][64]
    _Float16* h0 = &h0L[wid][0];
    _Float16* h1 = &h1L[wid][0];

    const int mOff = (lane >> 4) << 3;  // row offset of C/D VGPR block
    const int nLan = lane & 15;

    // Layer 0: [32x32] @ [32x64] -> relu -> h0 [32x64]
#pragma unroll
    for (int mt = 0; mt < 2; ++mt) {
        v16h a = load_frag_a(feat + mt * 16 * 32, 32, lane);
#pragma unroll
        for (int nt = 0; nt < 4; ++nt) {
            v16h b  = load_frag_b(w0T + nt * 16 * 32, 32, lane);
            v8f acc = {};
            acc = __builtin_amdgcn_wmma_f32_16x16x32_f16(false, a, false, b,
                                                         (short)0, acc, false, false);
            const int mrow = mt * 16 + mOff, ncol = nt * 16 + nLan;
#pragma unroll
            for (int r = 0; r < 8; ++r) {
                float x = acc[r];
                h0[(mrow + r) * 64 + ncol] = (_Float16)(x > 0.f ? x : 0.f);
            }
        }
    }
    asm volatile("s_wait_dscnt 0" ::: "memory");

    // Layer 1: [32x64] @ [64x64] -> relu -> h1 [32x64]
#pragma unroll
    for (int mt = 0; mt < 2; ++mt) {
        v16h a0 = load_frag_a(h0 + mt * 16 * 64 + 0,  64, lane);
        v16h a1 = load_frag_a(h0 + mt * 16 * 64 + 32, 64, lane);
#pragma unroll
        for (int nt = 0; nt < 4; ++nt) {
            v16h b0 = load_frag_b(w1T + nt * 16 * 64 + 0,  64, lane);
            v16h b1 = load_frag_b(w1T + nt * 16 * 64 + 32, 64, lane);
            v8f acc = {};
            acc = __builtin_amdgcn_wmma_f32_16x16x32_f16(false, a0, false, b0,
                                                         (short)0, acc, false, false);
            acc = __builtin_amdgcn_wmma_f32_16x16x32_f16(false, a1, false, b1,
                                                         (short)0, acc, false, false);
            const int mrow = mt * 16 + mOff, ncol = nt * 16 + nLan;
#pragma unroll
            for (int r = 0; r < 8; ++r) {
                float x = acc[r];
                h1[(mrow + r) * 64 + ncol] = (_Float16)(x > 0.f ? x : 0.f);
            }
        }
    }
    asm volatile("s_wait_dscnt 0" ::: "memory");

    // Layer 2: [32x64] @ [64x16(pad of 3)] -> sigmoid -> out [32x3]
#pragma unroll
    for (int mt = 0; mt < 2; ++mt) {
        v16h a0 = load_frag_a(h1 + mt * 16 * 64 + 0,  64, lane);
        v16h a1 = load_frag_a(h1 + mt * 16 * 64 + 32, 64, lane);
        v16h b0 = load_frag_b(w2T + 0,  64, lane);
        v16h b1 = load_frag_b(w2T + 32, 64, lane);
        v8f acc = {};
        acc = __builtin_amdgcn_wmma_f32_16x16x32_f16(false, a0, false, b0,
                                                     (short)0, acc, false, false);
        acc = __builtin_amdgcn_wmma_f32_16x16x32_f16(false, a1, false, b1,
                                                     (short)0, acc, false, false);
        if (nLan < 3) {
            const int mrow = ptBase + mt * 16 + mOff;
#pragma unroll
            for (int r = 0; r < 8; ++r) {
                int p = mrow + r;
                if (p < Npts) {
                    float x = acc[r];
                    out[p * 3 + nLan] = 1.0f / (1.0f + __expf(-x));
                }
            }
        }
    }
}

// ---------------------------------------------------------------------------
extern "C" void kernel_launch(void* const* d_in, const int* in_sizes, int n_in,
                              void* d_out, int out_size, void* d_ws, size_t ws_size,
                              hipStream_t stream) {
    const float* uv    = (const float*)d_in[0];
    const float* table = (const float*)d_in[1];
    const float* w0    = (const float*)d_in[2];
    const float* w1    = (const float*)d_in[3];
    const float* w2    = (const float*)d_in[4];
    float* out = (float*)d_out;
    const int Npts = in_sizes[0] / 2;

    _Float16* wsT = (_Float16*)d_ws;
    prep_weights<<<(7168 + 255) / 256, 256, 0, stream>>>(w0, w1, w2, wsT);

    LP lp;
    lp.dense_mask = 0;
    const double Bd = exp((log(2048.0) - log(16.0)) / 15.0);
    for (int l = 0; l < NLEV; ++l) {
        double sc = 16.0 * pow(Bd, (double)l) - 1.0;
        lp.scale[l] = (float)sc;
        long long r = (long long)ceil(sc) + 1;
        lp.res[l] = (unsigned)r;
        if (r * r <= (long long)TBL) lp.dense_mask |= (1u << l);
    }

    const int blocks = (Npts + 127) / 128;  // 128 threads = 4 waves, 32 pts/wave
    ngp_fused<<<blocks, 128, 0, stream>>>(uv, table, wsT, out, lp, Npts);
}
